// Block_10900626997991
// MI455X (gfx1250) — matrix operand order repaired and checked by reading, not compiled, not run on hardware
//
#include <hip/hip_runtime.h>
#include <hip/hip_bf16.h>
#include <math.h>

#define D_MODEL   1024
#define D_INNER   2048
#define D_STATE   64
#define D_CONV    4
#define HEADDIM   64
#define NHEADS    32                                   // D_INNER / HEADDIM
#define CONV_DIM  (D_INNER + 2 * D_STATE)              // 2176
#define D_IN_PROJ (2 * D_INNER + 2 * D_STATE + NHEADS) // 4256
#define ZX_N      4288                                 // D_IN_PROJ padded to mult. of 64
#define D_FF      (4 * D_MODEL)                        // 4096
#define BATCH     4
#define SEQ       4096
#define ROWS      (BATCH * SEQ)                        // 16384
#define LN_EPS    1e-5f

typedef __bf16 bf16;
typedef bf16  bf16x8  __attribute__((ext_vector_type(8)));
typedef bf16  bf16x16 __attribute__((ext_vector_type(16)));
typedef float f32x8   __attribute__((ext_vector_type(8)));

// ---------------------------------------------------------------------------
// f32 -> bf16 (grid-stride); rows >= n_real_rows are zero-filled (N padding)
// ---------------------------------------------------------------------------
__global__ __launch_bounds__(256)
void f32_to_bf16_kernel(const float* __restrict__ src, bf16* __restrict__ dst,
                        size_t n_real, size_t n_total) {
    size_t i = (size_t)blockIdx.x * blockDim.x + threadIdx.x;
    size_t stride = (size_t)gridDim.x * blockDim.x;
    for (; i < n_total; i += stride)
        dst[i] = (i < n_real) ? (bf16)src[i] : (bf16)0.0f;
}

// ---------------------------------------------------------------------------
// LayerNorm over D (=1024) columns, output bf16 (GEMM A operand)
// ---------------------------------------------------------------------------
__global__ __launch_bounds__(256)
void layernorm_bf16_kernel(const float* __restrict__ x, const float* __restrict__ w,
                           const float* __restrict__ b, bf16* __restrict__ out, int D) {
    const int row = blockIdx.x;
    const float* xr = x + (size_t)row * D;
    float s = 0.f, s2 = 0.f;
    for (int i = threadIdx.x; i < D; i += 256) {
        float v = xr[i];
        s += v; s2 += v * v;
    }
    #pragma unroll
    for (int off = 16; off > 0; off >>= 1) {
        s  += __shfl_down(s,  off, 32);
        s2 += __shfl_down(s2, off, 32);
    }
    __shared__ float rs[8], rs2[8];
    const int wave = threadIdx.x >> 5, lane = threadIdx.x & 31;
    if (lane == 0) { rs[wave] = s; rs2[wave] = s2; }
    __syncthreads();
    if (threadIdx.x == 0) {
        float a = 0.f, a2 = 0.f;
        for (int i = 0; i < 8; ++i) { a += rs[i]; a2 += rs2[i]; }
        rs[0] = a; rs2[0] = a2;
    }
    __syncthreads();
    const float mean = rs[0] / (float)D;
    const float var  = rs2[0] / (float)D - mean * mean;
    const float inv  = rsqrtf(var + LN_EPS);
    for (int i = threadIdx.x; i < D; i += 256) {
        float v = (xr[i] - mean) * inv * w[i] + b[i];
        out[(size_t)row * D + i] = (bf16)v;
    }
}

// ---------------------------------------------------------------------------
// Fragment loaders matching the CDNA5 WMMA VGPR layouts (ISA 7.12.2)
// A 16x32: lanes 0-15 row M=l, K {0..7,16..23}; lanes 16-31 K {8..15,24..31}
//   -> per lane: two 16B loads at +0 and +16 bf16 (base pre-offset by half*8)
// B 32x16: lane holds column n (l&15), same K order as A (pre-offset half*8)
//   -> per lane: two 16B loads at +0 and +16 bf16
// ---------------------------------------------------------------------------
__device__ __forceinline__ bf16x16 load_frag_pair(const bf16* __restrict__ p) {
    bf16x8 lo = *(const bf16x8*)(p);
    bf16x8 hi = *(const bf16x8*)(p + 16);
    return __builtin_shufflevector(lo, hi, 0, 1, 2, 3, 4, 5, 6, 7,
                                   8, 9, 10, 11, 12, 13, 14, 15);
}
__device__ __forceinline__ bf16x16 load_frag_contig(const bf16* __restrict__ p) {
    bf16x8 lo = *(const bf16x8*)(p);
    bf16x8 hi = *(const bf16x8*)(p + 8);
    return __builtin_shufflevector(lo, hi, 0, 1, 2, 3, 4, 5, 6, 7,
                                   8, 9, 10, 11, 12, 13, 14, 15);
}

// ---------------------------------------------------------------------------
// bf16 WMMA GEMM: C[M,N] = A[M,K] * W[N,K]^T  (+ fused epilogues)
//   MODE 0: store f32
//   MODE 1: +bias, exact GELU, store bf16 (MLP hidden)
//   MODE 2: +residual, store f32
//   MODE 3: +bias +residual, store f32
// Requirements: M % 256 == 0, N % 64 == 0, K % 32 == 0 (branch-free hot loop).
// Block: 256 thr = 8 waves stacked in M; wave tile 32x64 = 2x4 WMMA tiles.
// Per k-step a wave loads 64B A + 128B B for 8 WMMAs; B re-use across waves
// and blocks is served out of the 192MB L2 (all weights fit resident).
// ---------------------------------------------------------------------------
template <int MODE>
__global__ __launch_bounds__(256)
void gemm_wmma_bf16(const bf16* __restrict__ A, const bf16* __restrict__ W,
                    float* __restrict__ C, bf16* __restrict__ Cbf,
                    const float* __restrict__ bias, const float* __restrict__ res,
                    int M, int N, int K) {
    const int lane = threadIdx.x & 31;
    const int wave = threadIdx.x >> 5;
    const int half = lane >> 4;
    const int l15  = lane & 15;

    const int m0   = blockIdx.y * 256 + wave * 32;
    const int nblk = blockIdx.x * 64;

    f32x8 acc[2][4];
    #pragma unroll
    for (int r = 0; r < 2; ++r)
        #pragma unroll
        for (int j = 0; j < 4; ++j)
            #pragma unroll
            for (int i = 0; i < 8; ++i) acc[r][j][i] = 0.f;

    const bf16* aBase0 = A + (size_t)(m0 + l15) * K + half * 8;
    const bf16* aBase1 = aBase0 + (size_t)16 * K;
    const bf16* bBase  = W + (size_t)(nblk + l15) * K + half * 8;

    for (int kc = 0; kc < K; kc += 32) {
        const bf16x16 a0 = load_frag_pair(aBase0 + kc);
        const bf16x16 a1 = load_frag_pair(aBase1 + kc);
        #pragma unroll
        for (int j = 0; j < 4; ++j) {
            const bf16x16 bf = load_frag_pair(bBase + (size_t)j * 16 * K + kc);
            acc[0][j] = __builtin_amdgcn_wmma_f32_16x16x32_bf16(
                false, a0, false, bf, (short)0, acc[0][j], false, false);
            acc[1][j] = __builtin_amdgcn_wmma_f32_16x16x32_bf16(
                false, a1, false, bf, (short)0, acc[1][j], false, false);
        }
    }

    // Epilogue on the C/D layout: lane -> col nblk+j*16+l15, VGPR i -> row +half*8+i
    #pragma unroll
    for (int r = 0; r < 2; ++r) {
        #pragma unroll
        for (int j = 0; j < 4; ++j) {
            const int n = nblk + j * 16 + l15;
            #pragma unroll
            for (int i = 0; i < 8; ++i) {
                const int m = m0 + r * 16 + half * 8 + i;
                float v = acc[r][j][i];
                if (MODE == 1 || MODE == 3) v += bias[n];
                if (MODE == 1) {
                    v = 0.5f * v * (1.0f + erff(v * 0.7071067811865476f)); // exact GELU
                    Cbf[(size_t)m * N + n] = (bf16)v;
                } else {
                    if (MODE == 2 || MODE == 3) v += res[(size_t)m * N + n];
                    C[(size_t)m * N + n] = v;
                }
            }
        }
    }
}

// ---------------------------------------------------------------------------
// Depthwise causal conv(4) + SiLU over xBC slice, plus dt=softplus(.), dA=exp(dt*A)
// One block per token row. zxbcdt has leading dimension ZX_N (padded).
// ---------------------------------------------------------------------------
__global__ __launch_bounds__(256)
void conv_dt_prep_kernel(const float* __restrict__ zxbcdt,
                         const float* __restrict__ conv_w, const float* __restrict__ conv_b,
                         const float* __restrict__ dt_bias, const float* __restrict__ A_log,
                         float* __restrict__ xs, float* __restrict__ Bm, float* __restrict__ Cm,
                         float* __restrict__ dt, float* __restrict__ dA) {
    const int row = blockIdx.x;          // b*SEQ + l
    const int l   = row & (SEQ - 1);
    for (int c = threadIdx.x; c < CONV_DIM; c += 256) {
        float acc = conv_b[c];
        #pragma unroll
        for (int k = 0; k < D_CONV; ++k) {
            const int ll = l + k - (D_CONV - 1);
            if (ll >= 0)
                acc += conv_w[c * D_CONV + k] *
                       zxbcdt[(size_t)(row + k - (D_CONV - 1)) * ZX_N + D_INNER + c];
        }
        const float sv = acc / (1.0f + expf(-acc)); // SiLU
        if (c < D_INNER)                 xs[(size_t)row * D_INNER + c] = sv;
        else if (c < D_INNER + D_STATE)  Bm[(size_t)row * D_STATE + (c - D_INNER)] = sv;
        else                             Cm[(size_t)row * D_STATE + (c - D_INNER - D_STATE)] = sv;
    }
    if (threadIdx.x < NHEADS) {
        const int h = threadIdx.x;
        const float raw = zxbcdt[(size_t)row * ZX_N + (D_INNER + CONV_DIM) + h] + dt_bias[h];
        const float sp  = (raw > 20.0f) ? raw : log1pf(expf(raw)); // softplus
        const float An  = -expf(A_log[h]);
        dt[(size_t)row * NHEADS + h] = sp;
        dA[(size_t)row * NHEADS + h] = expf(sp * An);
    }
}

// ---------------------------------------------------------------------------
// Selective-state scan. One block per (batch, head); 64x64 state lives in
// registers (thread t -> p = t/4, n-quarter = t&3, 16 f32 each). B_t/C_t staged
// in LDS; y_p reduced across the 4 n-quarter lanes with __shfl_xor.
// ---------------------------------------------------------------------------
__global__ __launch_bounds__(256)
void ssm_scan_kernel(const float* __restrict__ xs, const float* __restrict__ Bm,
                     const float* __restrict__ Cm, const float* __restrict__ dt,
                     const float* __restrict__ dA, float* __restrict__ ys) {
    const int b  = blockIdx.x >> 5;  // 0..3
    const int hd = blockIdx.x & 31;  // head
    const int t  = threadIdx.x;
    const int p  = t >> 2;
    const int nq = t & 3;
    const int nb = nq * 16;

    float hstate[16];
    #pragma unroll
    for (int i = 0; i < 16; ++i) hstate[i] = 0.f;

    __shared__ float sB[D_STATE], sC[D_STATE];

    for (int l = 0; l < SEQ; ++l) {
        const size_t row = (size_t)b * SEQ + l;
        if (t < D_STATE)            sB[t]           = Bm[row * D_STATE + t];
        else if (t < 2 * D_STATE)   sC[t - D_STATE] = Cm[row * D_STATE + (t - D_STATE)];
        __syncthreads();

        const float dA_v = dA[row * NHEADS + hd];
        const float a    = dt[row * NHEADS + hd] * xs[row * D_INNER + hd * HEADDIM + p];
        float acc = 0.f;
        #pragma unroll
        for (int i = 0; i < 16; ++i) {
            const int n = nb + i;
            hstate[i] = dA_v * hstate[i] + a * sB[n];
            acc += hstate[i] * sC[n];
        }
        acc += __shfl_xor(acc, 1, 32);
        acc += __shfl_xor(acc, 2, 32);
        if (nq == 0) ys[row * D_INNER + hd * HEADDIM + p] = acc;
        __syncthreads();
    }
}

// ---------------------------------------------------------------------------
// y = (ys + D*xs) * silu(z); RMS-norm over D_INNER; store bf16 for out_proj A.
// ---------------------------------------------------------------------------
__global__ __launch_bounds__(256)
void gate_rmsnorm_kernel(const float* __restrict__ ys, const float* __restrict__ xs,
                         const float* __restrict__ zxbcdt, const float* __restrict__ Dp,
                         const float* __restrict__ norm_w, bf16* __restrict__ ybf) {
    const int row = blockIdx.x;
    __shared__ float buf[D_INNER];
    float s2 = 0.f;
    for (int c = threadIdx.x; c < D_INNER; c += 256) {
        const int h = c >> 6;
        float y = ys[(size_t)row * D_INNER + c] + Dp[h] * xs[(size_t)row * D_INNER + c];
        const float z = zxbcdt[(size_t)row * ZX_N + c];
        y *= z / (1.0f + expf(-z));
        buf[c] = y;
        s2 += y * y;
    }
    #pragma unroll
    for (int off = 16; off > 0; off >>= 1) s2 += __shfl_down(s2, off, 32);
    __shared__ float rs[8];
    if ((threadIdx.x & 31) == 0) rs[threadIdx.x >> 5] = s2;
    __syncthreads();
    if (threadIdx.x == 0) { float a = 0.f; for (int i = 0; i < 8; ++i) a += rs[i]; rs[0] = a; }
    __syncthreads();
    const float inv = rsqrtf(rs[0] / (float)D_INNER + LN_EPS);
    for (int c = threadIdx.x; c < D_INNER; c += 256)
        ybf[(size_t)row * D_INNER + c] = (bf16)(buf[c] * inv * norm_w[c]);
}

// ---------------------------------------------------------------------------
extern "C" void kernel_launch(void* const* d_in, const int* in_sizes, int n_in,
                              void* d_out, int out_size, void* d_ws, size_t ws_size,
                              hipStream_t stream) {
    (void)in_sizes; (void)n_in; (void)out_size; (void)ws_size;

    const float* x          = (const float*)d_in[0];
    const float* ln1_w      = (const float*)d_in[1];
    const float* ln1_b      = (const float*)d_in[2];
    const float* ln2_w      = (const float*)d_in[3];
    const float* ln2_b      = (const float*)d_in[4];
    const float* in_proj_w  = (const float*)d_in[5];
    const float* conv_w     = (const float*)d_in[6];
    const float* conv_b     = (const float*)d_in[7];
    const float* dt_bias    = (const float*)d_in[8];
    const float* A_log      = (const float*)d_in[9];
    const float* D_param    = (const float*)d_in[10];
    const float* norm_w     = (const float*)d_in[11];
    const float* out_proj_w = (const float*)d_in[12];
    const float* fc_w       = (const float*)d_in[13];
    const float* fc_b       = (const float*)d_in[14];
    const float* proj_w     = (const float*)d_in[15];
    const float* proj_b     = (const float*)d_in[16];

    // --- workspace bump allocator (256B aligned) ---
    char* ws = (char*)d_ws;
    size_t off = 0;
    auto alloc = [&](size_t bytes) -> char* {
        char* p = ws + off;
        off += (bytes + 255) & ~(size_t)255;
        return p;
    };
    bf16*  lnbf = (bf16*)alloc((size_t)ROWS * D_MODEL * 2);          // ln1 / ln2 (reused)
    bf16*  w_in = (bf16*)alloc((size_t)ZX_N * D_MODEL * 2);          // padded rows -> 0
    bf16*  w_op = (bf16*)alloc((size_t)D_MODEL * D_INNER * 2);
    bf16*  w_fc = (bf16*)alloc((size_t)D_FF * D_MODEL * 2);
    bf16*  w_pr = (bf16*)alloc((size_t)D_MODEL * D_FF * 2);
    float* zx   = (float*)alloc((size_t)ROWS * ZX_N * 4);            // zxbcdt (ld = ZX_N)
    float* xs   = (float*)alloc((size_t)ROWS * D_INNER * 4);
    float* Bm   = (float*)alloc((size_t)ROWS * D_STATE * 4);
    float* Cm   = (float*)alloc((size_t)ROWS * D_STATE * 4);
    float* dtb  = (float*)alloc((size_t)ROWS * NHEADS * 4);
    float* dAb  = (float*)alloc((size_t)ROWS * NHEADS * 4);
    float* ysb  = (float*)alloc((size_t)ROWS * D_INNER * 4);
    bf16*  ybf  = (bf16*)alloc((size_t)ROWS * D_INNER * 2);
    float* x1   = (float*)alloc((size_t)ROWS * D_MODEL * 4);
    bf16*  hbf  = (bf16*)zx;  // overlay: zxbcdt dead after gate kernel

    // 1) weights -> bf16 (in_proj padded 4256 -> 4288 rows of zeros)
    f32_to_bf16_kernel<<<2048, 256, 0, stream>>>(in_proj_w,  w_in,
        (size_t)D_IN_PROJ * D_MODEL, (size_t)ZX_N * D_MODEL);
    f32_to_bf16_kernel<<<2048, 256, 0, stream>>>(out_proj_w, w_op,
        (size_t)D_MODEL * D_INNER, (size_t)D_MODEL * D_INNER);
    f32_to_bf16_kernel<<<2048, 256, 0, stream>>>(fc_w,       w_fc,
        (size_t)D_FF * D_MODEL, (size_t)D_FF * D_MODEL);
    f32_to_bf16_kernel<<<2048, 256, 0, stream>>>(proj_w,     w_pr,
        (size_t)D_MODEL * D_FF, (size_t)D_MODEL * D_FF);

    // 2) ln1(x) -> bf16
    layernorm_bf16_kernel<<<ROWS, 256, 0, stream>>>(x, ln1_w, ln1_b, lnbf, D_MODEL);

    // 3) zxbcdt = ln1 @ in_proj^T   (N padded to 4288; pad cols compute to 0)
    dim3 blk(256);
    dim3 g1(ZX_N / 64, ROWS / 256);
    gemm_wmma_bf16<0><<<g1, blk, 0, stream>>>(lnbf, w_in, zx, nullptr, nullptr, nullptr,
                                              ROWS, ZX_N, D_MODEL);

    // 4) conv + silu + dt/dA
    conv_dt_prep_kernel<<<ROWS, 256, 0, stream>>>(zx, conv_w, conv_b, dt_bias, A_log,
                                                  xs, Bm, Cm, dtb, dAb);

    // 5) sequential selective scan (parallel over B*NHEADS = 128 blocks)
    ssm_scan_kernel<<<BATCH * NHEADS, 256, 0, stream>>>(xs, Bm, Cm, dtb, dAb, ysb);

    // 6) gate + RMS norm -> bf16
    gate_rmsnorm_kernel<<<ROWS, 256, 0, stream>>>(ysb, xs, zx, D_param, norm_w, ybf);

    // 7) x1 = x + y @ out_proj^T
    dim3 g2(D_MODEL / 64, ROWS / 256);
    gemm_wmma_bf16<2><<<g2, blk, 0, stream>>>(ybf, w_op, x1, nullptr, nullptr, x,
                                              ROWS, D_MODEL, D_INNER);

    // 8) ln2(x1) -> bf16
    layernorm_bf16_kernel<<<ROWS, 256, 0, stream>>>(x1, ln2_w, ln2_b, lnbf, D_MODEL);

    // 9) h = gelu(ln2 @ fc^T + fc_b) -> bf16 (overlays zxbcdt)
    dim3 g3(D_FF / 64, ROWS / 256);
    gemm_wmma_bf16<1><<<g3, blk, 0, stream>>>(lnbf, w_fc, nullptr, hbf, fc_b, nullptr,
                                              ROWS, D_FF, D_MODEL);

    // 10) out = x1 + h @ proj^T + proj_b
    dim3 g4(D_MODEL / 64, ROWS / 256);
    gemm_wmma_bf16<3><<<g4, blk, 0, stream>>>(hbf, w_pr, (float*)d_out, nullptr, proj_b, x1,
                                              ROWS, D_MODEL, D_FF);
}
